// MyAttentiveLSTM_13700945674877
// MI455X (gfx1250) — matrix operand order repaired and checked by reading, not compiled, not run on hardware
//
#include <hip/hip_runtime.h>
#include <math.h>

typedef __attribute__((ext_vector_type(16))) __bf16 v16bf;
typedef __attribute__((ext_vector_type(8)))  float  v8f;

struct alignas(16) F4 { float x, y, z, w; };

union ABFrag {
  F4    f4[2];
  v16bf v;
};

__device__ __forceinline__ unsigned short f2bf(float f) {
  unsigned u = __builtin_bit_cast(unsigned, f);
  unsigned r = u + 0x7FFFu + ((u >> 16) & 1u);   // round-to-nearest-even
  return (unsigned short)(r >> 16);
}

#define BATCH 8
#define CH    512
#define HW    256          // 16*16
#define MTOT  2048         // BATCH*HW
#define NTOT  2048         // 4 gates * 512

// ---------------------------------------------------------------------------
// Fold WQ (131072x256) with WK (512) -> wqk[hw=256][t=256]
// ---------------------------------------------------------------------------
__global__ void k_wqk(const float* __restrict__ WQ, const float* __restrict__ WK,
                      float* __restrict__ wqk) {
  int idx = blockIdx.x * 256 + threadIdx.x;      // 65536 total
  int hw = idx >> 8, t = idx & 255;
  const float* col = WQ + (size_t)hw * 512u * 256u + t;
  float s = 0.f;
  for (int e = 0; e < 512; ++e) s += WK[e] * col[(size_t)e * 256u];
  wqk[hw * 256 + t] = s;
}

// ---------------------------------------------------------------------------
// myA[b,hw] = 1 + softmax_H((txt.wqk)/sqrt(512)) * (WV.Wsum_w) + Wsum_b
// ---------------------------------------------------------------------------
__global__ void k_myA(const float* __restrict__ txt, const float* __restrict__ wqk,
                      const float* __restrict__ WV, const float* __restrict__ Wsw,
                      const float* __restrict__ Wsb, float* __restrict__ myA) {
  int b = blockIdx.x;            // 8 blocks
  int hw = threadIdx.x;          // 256 threads
  const float* t2 = txt + b * 256;
  float q = 0.f;
  for (int t = 0; t < 256; ++t) q += t2[t] * wqk[hw * 256 + t];
  float score = q * (1.0f / sqrtf(512.0f));
  __shared__ float sc[256];
  sc[hw] = score;
  __syncthreads();
  int w = hw & 15;
  float mx = -1e30f;
  for (int h = 0; h < 16; ++h) mx = fmaxf(mx, sc[h * 16 + w]);
  float den = 0.f;
  for (int h = 0; h < 16; ++h) den += __expf(sc[h * 16 + w] - mx);
  float attn = __expf(score - mx) / den;
  float cVW = 0.f;
  for (int e = 0; e < 512; ++e) cVW += WV[e] * Wsw[e];
  myA[b * 256 + hw] = 1.0f + attn * cVW + Wsb[0];
}

// ---------------------------------------------------------------------------
// x (NCHW f32) -> xbf (NHWC bf16), c0 (NHWC f32), AMbf = myA*x (NHWC bf16)
// ---------------------------------------------------------------------------
__global__ void k_prep(const float* __restrict__ x, const float* __restrict__ myA,
                       unsigned short* __restrict__ xbf, float* __restrict__ cst,
                       unsigned short* __restrict__ ambf) {
  int idx = blockIdx.x * 256 + threadIdx.x;      // MTOT*CH = 1M
  int m = idx >> 9, c = idx & 511;
  int b = m >> 8, hw = m & 255;
  float xv = x[((size_t)b * 512 + c) * 256 + hw];
  xbf[idx]  = f2bf(xv);
  cst[idx]  = xv;
  ambf[idx] = f2bf(xv * myA[m]);
}

// ---------------------------------------------------------------------------
// Pack OIHW f32 [512][512][3][3] -> bf16 [tap=9][NTOT][cin=512] at gate slot
// ---------------------------------------------------------------------------
__global__ void k_pack_w(const float* __restrict__ W, unsigned short* __restrict__ wt,
                         int slot) {
  int idx = blockIdx.x * 256 + threadIdx.x;      // 9*512*512
  if (idx >= 9 * 512 * 512) return;
  int tap = idx / (512 * 512);
  int rem = idx - tap * 512 * 512;
  int co = rem >> 9, ci = rem & 511;
  wt[((size_t)tap * NTOT + slot * 512 + co) * 512 + ci] =
      f2bf(W[((size_t)co * 512 + ci) * 9 + tap]);
}

__global__ void k_pack_b(const float* __restrict__ b0, const float* __restrict__ b1,
                         const float* __restrict__ b2, const float* __restrict__ b3,
                         float* __restrict__ dst) {
  int i = blockIdx.x * 256 + threadIdx.x;        // 2048
  if (i >= 2048) return;
  int s = i >> 9, k = i & 511;
  const float* src = (s == 0) ? b0 : (s == 1) ? b1 : (s == 2) ? b2 : b3;
  dst[i] = src[k];
}

// ---------------------------------------------------------------------------
// Implicit-GEMM 3x3 conv via V_WMMA_F32_16X16X32_BF16, 2Mx2N register blocking.
//   act: bf16 NHWC [8][16][16][512], wt: bf16 [9][Ntot][512], out: f32 [2048][Ntot]
// Each wave: M-tiles (mg, mg+64)  -> same yy, batches b and b+4 (same padding
// predicate, wave-uniform skip); N-tiles (nt, nt+64). 4 accumulators,
// 8 b128 loads : 4 WMMA per K-step.
// ---------------------------------------------------------------------------
__global__ void k_conv_wmma(const unsigned short* __restrict__ act,
                            const unsigned short* __restrict__ wt,
                            const float* __restrict__ bias,
                            float* __restrict__ out, int Ntot) {
  const int wave = threadIdx.x >> 5;             // blockDim = 128 -> 4 waves
  const int lane = threadIdx.x & 31;
  const int mg   = blockIdx.x;                   // 0..63 : M-tile pair (mg, mg+64)
  const int nt0  = blockIdx.y * 4 + wave;        // 0..63 : N-tile pair (nt0, nt0+64)
  const int b0 = mg >> 4;                        // batches 0..3 (pair batch = b0+4)
  const int yy = mg & 15;
  const int xx = lane & 15;                      // output x == A-row M, and N-col
  const int hi = lane >> 4;                      // lane half selects K interleave
  const int klo = hi ? 8 : 0;
  const int n0 = nt0 * 16 + xx;
  const int n1 = n0 + 1024;

  v8f acc00 = {0.f, 0.f, 0.f, 0.f, 0.f, 0.f, 0.f, 0.f};
  v8f acc01 = acc00, acc10 = acc00, acc11 = acc00;

  for (int tap = 0; tap < 9; ++tap) {
    const int dy = tap / 3 - 1, dx = tap % 3 - 1;
    const int py = yy + dy;                      // wave-uniform
    if (py < 0 || py > 15) continue;             // uniform branch: EXEC stays full
    const int px = xx + dx;
    const bool inb = (px >= 0) && (px <= 15);
    const unsigned short* a0row =
        act + (((size_t)(b0 * 16 + py) * 16 + (inb ? px : 0)) * 512);
    const unsigned short* a1row = a0row + (size_t)4 * 256 * 512;  // batch b0+4
    const unsigned short* w0row = wt + ((size_t)tap * Ntot + n0) * 512 + hi * 8;
    const unsigned short* w1row = w0row + (size_t)1024 * 512;
#pragma unroll 2
    for (int kb = 0; kb < 16; ++kb) {
      const int cb = kb * 32;
      ABFrag a0, a1, bb0, bb1;
      if (inb) {
        a0.f4[0] = *(const F4*)(a0row + cb + klo);        // K = cb+klo    ..+7
        a0.f4[1] = *(const F4*)(a0row + cb + klo + 16);   // K = cb+klo+16 ..+7
        a1.f4[0] = *(const F4*)(a1row + cb + klo);
        a1.f4[1] = *(const F4*)(a1row + cb + klo + 16);
      } else {                                            // conv zero padding
        a0.f4[0] = F4{0.f, 0.f, 0.f, 0.f}; a0.f4[1] = a0.f4[0];
        a1.f4[0] = a0.f4[0];               a1.f4[1] = a0.f4[0];
      }
      bb0.f4[0] = *(const F4*)(w0row + cb);               // K = cb+hi*8   ..+7
      bb0.f4[1] = *(const F4*)(w0row + cb + 16);          // K = cb+hi*8+16..+7
      bb1.f4[0] = *(const F4*)(w1row + cb);
      bb1.f4[1] = *(const F4*)(w1row + cb + 16);
      acc00 = __builtin_amdgcn_wmma_f32_16x16x32_bf16(
          false, a0.v, false, bb0.v, (short)0, acc00, false, false);
      acc01 = __builtin_amdgcn_wmma_f32_16x16x32_bf16(
          false, a0.v, false, bb1.v, (short)0, acc01, false, false);
      acc10 = __builtin_amdgcn_wmma_f32_16x16x32_bf16(
          false, a1.v, false, bb0.v, (short)0, acc10, false, false);
      acc11 = __builtin_amdgcn_wmma_f32_16x16x32_bf16(
          false, a1.v, false, bb1.v, (short)0, acc11, false, false);
    }
  }

  const float bv0 = bias[n0];
  const float bv1 = bias[n1];
#pragma unroll
  for (int v = 0; v < 8; ++v) {                  // D: VGPR v -> row M = v + hi*8
    const int m0 = mg * 16 + v + hi * 8;         // tiles mg and mg+64
    const int m1 = m0 + 1024;
    out[(size_t)m0 * Ntot + n0] = acc00[v] + bv0;
    out[(size_t)m0 * Ntot + n1] = acc01[v] + bv1;
    out[(size_t)m1 * Ntot + n0] = acc10[v] + bv0;
    out[(size_t)m1 * Ntot + n1] = acc11[v] + bv1;
  }
}

// ---------------------------------------------------------------------------
// LSTM gates: i,f,c,o from Z (W*AM, const) + G (U*h); update c, h.
// ---------------------------------------------------------------------------
__global__ void k_gate(const float* __restrict__ Z, const float* __restrict__ G,
                       float* __restrict__ cst, unsigned short* __restrict__ hbf,
                       float* __restrict__ out, int last) {
  int idx = blockIdx.x * 256 + threadIdx.x;      // 1M
  int m = idx >> 9, c = idx & 511;
  size_t base = (size_t)m * NTOT;
  float zi = Z[base + c]        + G[base + c];
  float zf = Z[base + 512 + c]  + G[base + 512 + c];
  float zc = Z[base + 1024 + c] + G[base + 1024 + c];
  float zo = Z[base + 1536 + c] + G[base + 1536 + c];
  float gi = 1.f / (1.f + __expf(-zi));
  float gf = 1.f / (1.f + __expf(-zf));
  float go = 1.f / (1.f + __expf(-zo));
  float gg = tanhf(zc);
  float cv = gg * gi + gf * cst[idx];
  cst[idx] = cv;
  float h = tanhf(cv) * go;
  hbf[idx] = f2bf(h);
  if (last) {
    int b = m >> 8, yy = (m >> 4) & 15, px = m & 15;
    out[(((size_t)b * 512 + c) * 16 + yy) * 16 + px] = h;
  }
}

// ---------------------------------------------------------------------------
extern "C" void kernel_launch(void* const* d_in, const int* in_sizes, int n_in,
                              void* d_out, int out_size, void* d_ws, size_t ws_size,
                              hipStream_t stream) {
  (void)in_sizes; (void)n_in; (void)out_size; (void)ws_size;
  const float* x    = (const float*)d_in[0];
  const float* txt  = (const float*)d_in[1];
  // d_in[2..6] = Wa/Ua/Va : dead code (softmax over singleton channel dim == 1)
  const float* Wi_w = (const float*)d_in[7];   const float* Wi_b = (const float*)d_in[8];
  const float* Ui_w = (const float*)d_in[9];   const float* Ui_b = (const float*)d_in[10];
  const float* Wf_w = (const float*)d_in[11];  const float* Wf_b = (const float*)d_in[12];
  const float* Uf_w = (const float*)d_in[13];  const float* Uf_b = (const float*)d_in[14];
  const float* Wc_w = (const float*)d_in[15];  const float* Wc_b = (const float*)d_in[16];
  const float* Uc_w = (const float*)d_in[17];  const float* Uc_b = (const float*)d_in[18];
  const float* Wo_w = (const float*)d_in[19];  const float* Wo_b = (const float*)d_in[20];
  const float* Uo_w = (const float*)d_in[21];  const float* Uo_b = (const float*)d_in[22];
  const float* WQ   = (const float*)d_in[23];
  const float* WK   = (const float*)d_in[24];
  const float* WV   = (const float*)d_in[25];
  const float* Wsw  = (const float*)d_in[26];
  const float* Wsb  = (const float*)d_in[27];
  float* out = (float*)d_out;

  // carve workspace (d_ws base is 256B aligned)
  char* ws = (char*)d_ws;
  auto carve = [&](size_t bytes) {
    char* p = ws;
    ws += (bytes + 255) & ~(size_t)255;
    return p;
  };
  float*          wqk  = (float*)carve(256 * 256 * 4);
  float*          myA  = (float*)carve(MTOT * 4);
  unsigned short* xbf  = (unsigned short*)carve((size_t)MTOT * CH * 2);
  unsigned short* ambf = (unsigned short*)carve((size_t)MTOT * CH * 2);
  unsigned short* hbf  = (unsigned short*)carve((size_t)MTOT * CH * 2);
  float*          cst  = (float*)carve((size_t)MTOT * CH * 4);
  unsigned short* WtW  = (unsigned short*)carve((size_t)9 * NTOT * 512 * 2);
  unsigned short* WtU  = (unsigned short*)carve((size_t)9 * NTOT * 512 * 2);
  float*          bW   = (float*)carve(NTOT * 4);
  float*          bU   = (float*)carve(NTOT * 4);
  float*          Zbuf = (float*)carve((size_t)MTOT * NTOT * 4);
  float*          Gbuf = (float*)carve((size_t)MTOT * NTOT * 4);

  // 1) fold WQ with WK (one 134MB streaming pass), text attention field
  k_wqk<<<256, 256, 0, stream>>>(WQ, WK, wqk);
  k_myA<<<BATCH, 256, 0, stream>>>(txt, wqk, WV, Wsw, Wsb, myA);
  k_prep<<<4096, 256, 0, stream>>>(x, myA, xbf, cst, ambf);

  // 2) pack gate weights bf16 [tap][cout][cin]; slots: i,f,c,o
  const int wpgrid = (9 * 512 * 512 + 255) / 256;
  k_pack_w<<<wpgrid, 256, 0, stream>>>(Wi_w, WtW, 0);
  k_pack_w<<<wpgrid, 256, 0, stream>>>(Wf_w, WtW, 1);
  k_pack_w<<<wpgrid, 256, 0, stream>>>(Wc_w, WtW, 2);
  k_pack_w<<<wpgrid, 256, 0, stream>>>(Wo_w, WtW, 3);
  k_pack_w<<<wpgrid, 256, 0, stream>>>(Ui_w, WtU, 0);
  k_pack_w<<<wpgrid, 256, 0, stream>>>(Uf_w, WtU, 1);
  k_pack_w<<<wpgrid, 256, 0, stream>>>(Uc_w, WtU, 2);
  k_pack_w<<<wpgrid, 256, 0, stream>>>(Uo_w, WtU, 3);
  k_pack_b<<<8, 256, 0, stream>>>(Wi_b, Wf_b, Wc_b, Wo_b, bW);
  k_pack_b<<<8, 256, 0, stream>>>(Ui_b, Uf_b, Uc_b, Uo_b, bU);

  // 3) Z = conv(AM, [Wi|Wf|Wc|Wo]) + bW  — timestep-invariant, computed once
  dim3 cgrid(64, 16);                            // M-pairs x N-pair groups
  k_conv_wmma<<<cgrid, 128, 0, stream>>>(ambf, WtW, bW, Zbuf, NTOT);

  // 4) 4 timesteps: G = conv(h, [Ui|Uf|Uc|Uo]) + bU, then gate update
  for (int ts = 0; ts < 4; ++ts) {
    const unsigned short* hin = (ts == 0) ? xbf : hbf;
    k_conv_wmma<<<cgrid, 128, 0, stream>>>(hin, WtU, bU, Gbuf, NTOT);
    k_gate<<<4096, 256, 0, stream>>>(Zbuf, Gbuf, cst, hbf, out, (ts == 3) ? 1 : 0);
  }
}
